// TemporalMambaStack_17781164605844
// MI455X (gfx1250) — matrix operand order repaired and checked by reading, not compiled, not run on hardware
//
#include <hip/hip_runtime.h>
#include <hip/hip_bf16.h>
#include <cstdint>
#include <cstddef>

// Problem constants (match reference)
#define DIMV   512
#define NLAY   2
#define DSTV   16
#define DCV    4
#define DINV   1024
#define DTRV   32
#define BSZV   4
#define SEQV   1024
#define MROWS  (BSZV * SEQV)   // 4096
#define XDBW   (DTRV + 2 * DSTV)  // 64

typedef __attribute__((ext_vector_type(16))) __bf16 v16bf;
typedef __attribute__((ext_vector_type(8)))  __bf16 v8bf;
typedef __attribute__((ext_vector_type(8)))  float  v8f;

__device__ __forceinline__ __bf16 f2bf(float f) {
  union { float f; uint32_t u; } in; in.f = f;
  uint32_t r = in.u + 0x7FFFu + ((in.u >> 16) & 1u);   // round-to-nearest-even
  union { uint16_t s; __bf16 b; } out; out.s = (uint16_t)(r >> 16);
  return out.b;
}

__device__ __forceinline__ float sigmoid_f(float v) {
  return 1.0f / (1.0f + __expf(-v));
}

// ---------------------------------------------------------------------------
// Elementwise f32 -> bf16 convert (4 elements / thread, vectorized load)
// ---------------------------------------------------------------------------
__global__ __launch_bounds__(256) void cvt_bf16_kernel(
    const float* __restrict__ src, __bf16* __restrict__ dst, int n)
{
  const int i = (blockIdx.x * blockDim.x + threadIdx.x) * 4;
  if (i >= n) return;
  const float4 v = *reinterpret_cast<const float4*>(src + i);
  dst[i + 0] = f2bf(v.x);
  dst[i + 1] = f2bf(v.y);
  dst[i + 2] = f2bf(v.z);
  dst[i + 3] = f2bf(v.w);
}

// ---------------------------------------------------------------------------
// C[M,N] = A[M,K](lda,bf16) * W[N,K]^T(ldw,bf16), f32 accumulate via WMMA.
// Wave tile = 32x32 (4 accumulators, A/B fragment reuse).
// Optional f32 output C, optional bf16 output Cbf (same ldc).
// act==1: softplus(val + bias[col]).
// ---------------------------------------------------------------------------
__global__ __launch_bounds__(256) void gemm_wmma_bf16(
    const __bf16* __restrict__ A, int lda,
    const __bf16* __restrict__ W, int ldw,
    float* __restrict__ C, __bf16* __restrict__ Cbf, int ldc,
    int M, int N, int K,
    const float* __restrict__ bias, int act)
{
  const int lane   = threadIdx.x & 31;
  const int wid    = blockIdx.x * (blockDim.x >> 5) + (threadIdx.x >> 5);
  const int tilesN = N >> 5;
  const int tilesM = M >> 5;
  if (wid >= tilesM * tilesN) return;           // whole wave exits together
  const int tm   = (wid / tilesN) << 5;
  const int tn   = (wid % tilesN) << 5;
  const int half = lane >> 4;
  const int l15  = lane & 15;

  const __bf16* Ap0 = A + (size_t)(tm + l15) * lda;
  const __bf16* Ap1 = A + (size_t)(tm + 16 + l15) * lda;
  const __bf16* Wp0 = W + (size_t)(tn + l15) * ldw;
  const __bf16* Wp1 = W + (size_t)(tn + 16 + l15) * ldw;

  v8f acc00 = {}, acc01 = {}, acc10 = {}, acc11 = {};

  for (int k0 = 0; k0 < K; k0 += 32) {
    // A fragment (16x32 bf16): k = 16*half + e: two b128 runs
    const v8bf a0lo = *reinterpret_cast<const v8bf*>(Ap0 + k0 + 16 * half);
    const v8bf a0hi = *reinterpret_cast<const v8bf*>(Ap0 + k0 + 16 * half + 8);
    const v8bf a1lo = *reinterpret_cast<const v8bf*>(Ap1 + k0 + 16 * half);
    const v8bf a1hi = *reinterpret_cast<const v8bf*>(Ap1 + k0 + 16 * half + 8);
    // B fragment (32x16 bf16): k = 16*half + e: one contiguous 32B run
    const v16bf b0 = *reinterpret_cast<const v16bf*>(Wp0 + k0 + 16 * half);
    const v16bf b1 = *reinterpret_cast<const v16bf*>(Wp1 + k0 + 16 * half);

    if (k0 + 32 < K) {                 // global_prefetch_b8 for next k-step
      __builtin_prefetch(Ap0 + k0 + 32, 0, 0);
      __builtin_prefetch(Ap1 + k0 + 32, 0, 0);
      __builtin_prefetch(Wp0 + k0 + 32, 0, 0);
      __builtin_prefetch(Wp1 + k0 + 32, 0, 0);
    }

    const v16bf a0 = __builtin_shufflevector(a0lo, a0hi,
        0, 1, 2, 3, 4, 5, 6, 7, 8, 9, 10, 11, 12, 13, 14, 15);
    const v16bf a1 = __builtin_shufflevector(a1lo, a1hi,
        0, 1, 2, 3, 4, 5, 6, 7, 8, 9, 10, 11, 12, 13, 14, 15);

    acc00 = __builtin_amdgcn_wmma_f32_16x16x32_bf16(false, a0, false, b0, (short)0, acc00, false, false);
    acc01 = __builtin_amdgcn_wmma_f32_16x16x32_bf16(false, a0, false, b1, (short)0, acc01, false, false);
    acc10 = __builtin_amdgcn_wmma_f32_16x16x32_bf16(false, a1, false, b0, (short)0, acc10, false, false);
    acc11 = __builtin_amdgcn_wmma_f32_16x16x32_bf16(false, a1, false, b1, (short)0, acc11, false, false);
  }

  // Epilogue: C/D layout: VGPR r -> row r + 8*half, col = lane&15
#pragma unroll
  for (int tj = 0; tj < 2; ++tj) {
    const int col = tn + 16 * tj + l15;
    const float bv = bias ? bias[col] : 0.0f;
#pragma unroll
    for (int ti = 0; ti < 2; ++ti) {
      const v8f acc = (ti == 0) ? (tj == 0 ? acc00 : acc01)
                                : (tj == 0 ? acc10 : acc11);
#pragma unroll
      for (int r = 0; r < 8; ++r) {
        const int row = tm + 16 * ti + r + 8 * half;
        float v = acc[r] + bv;
        if (act == 1) v = (v > 20.0f) ? v : log1pf(__expf(v));
        if (C)   C[(size_t)row * ldc + col]   = v;
        if (Cbf) Cbf[(size_t)row * ldc + col] = f2bf(v);
      }
    }
  }
}

// ---------------------------------------------------------------------------
// Depthwise causal conv (DC=4 taps, left zero-pad) + bias + SiLU.
// Writes f32 u (for the scan) and bf16 u (for the next GEMM).
// ---------------------------------------------------------------------------
__global__ __launch_bounds__(256) void conv_silu_kernel(
    const float* __restrict__ xi, const float* __restrict__ cw,
    const float* __restrict__ cb, float* __restrict__ u,
    __bf16* __restrict__ ubf)
{
  const int idx = blockIdx.x * blockDim.x + threadIdx.x;
  const int d = idx % DINV;
  const int t = (idx / DINV) % SEQV;
  const int b = idx / (DINV * SEQV);
  const float* base = xi + ((size_t)b * SEQV) * DINV + d;
  float v = cb[d];
#pragma unroll
  for (int j = 0; j < DCV; ++j) {
    const int tt = t - (DCV - 1) + j;
    if (tt >= 0) v += base[(size_t)tt * DINV] * cw[d * DCV + j];
  }
  const float s = v * sigmoid_f(v);
  u[idx]   = s;
  ubf[idx] = f2bf(s);
}

// ---------------------------------------------------------------------------
// Selective scan + skip + gate. One lane per (b,d) channel; 16-wide state in
// registers; per-t B/C staged through LDS. Emits gated output directly in bf16.
// ---------------------------------------------------------------------------
__global__ __launch_bounds__(256) void scan_gate_kernel(
    const float* __restrict__ delta, const float* __restrict__ u,
    const float* __restrict__ xdb, const float* __restrict__ A_log,
    const float* __restrict__ Dv, const float* __restrict__ zb,
    __bf16* __restrict__ ygbf)
{
  __shared__ float sBC[2 * DSTV];
  const int gid = blockIdx.x * blockDim.x + threadIdx.x;
  const int b = gid / DINV;          // all threads in a block share the same b
  const int d = gid % DINV;

  float Arow[DSTV];
#pragma unroll
  for (int s = 0; s < DSTV; ++s) Arow[s] = -__expf(A_log[d * DSTV + s]);
  const float Dd = Dv[d];

  float st[DSTV];
#pragma unroll
  for (int s = 0; s < DSTV; ++s) st[s] = 0.0f;

  for (int t = 0; t < SEQV; ++t) {
    const size_t rb = (size_t)b * SEQV + t;
    __syncthreads();
    if (threadIdx.x < 2 * DSTV)
      sBC[threadIdx.x] = xdb[rb * XDBW + DTRV + threadIdx.x];
    __syncthreads();

    const float del = delta[rb * DINV + d];
    const float uu  = u[rb * DINV + d];
    const float du  = del * uu;
    float y = 0.0f;
#pragma unroll
    for (int s = 0; s < DSTV; ++s) {
      const float dA = __expf(del * Arow[s]);
      st[s] = st[s] * dA + du * sBC[s];
      y += st[s] * sBC[DSTV + s];
    }
    y += uu * Dd;
    const float zz = zb[rb * DINV + d];
    y *= zz * sigmoid_f(zz);                 // y * silu(z)
    ygbf[rb * DINV + d] = f2bf(y);
  }
}

// ---------------------------------------------------------------------------
extern "C" void kernel_launch(void* const* d_in, const int* in_sizes, int n_in,
                              void* d_out, int out_size, void* d_ws, size_t ws_size,
                              hipStream_t stream)
{
  const float* x      = (const float*)d_in[0];
  const float* W_in   = (const float*)d_in[1];
  const float* conv_w = (const float*)d_in[2];
  const float* conv_b = (const float*)d_in[3];
  const float* W_x    = (const float*)d_in[4];
  const float* W_dt   = (const float*)d_in[5];
  const float* b_dt   = (const float*)d_in[6];
  const float* A_log  = (const float*)d_in[7];
  const float* Dv     = (const float*)d_in[8];
  const float* W_out  = (const float*)d_in[9];

  // Workspace carve-out (256B-aligned chunks)
  char* wp = (char*)d_ws;
  auto alloc_f32 = [&](size_t n) {
    float* r = (float*)wp; wp += ((n * 4 + 255) / 256) * 256; return r;
  };
  auto alloc_bf = [&](size_t n) {
    __bf16* r = (__bf16*)wp; wp += ((n * 2 + 255) / 256) * 256; return r;
  };

  const size_t SZ_BLD = (size_t)MROWS * DINV;      // 4M elements
  float*  xi     = alloc_f32(SZ_BLD);
  float*  zbuf   = alloc_f32(SZ_BLD);
  float*  ubuf   = alloc_f32(SZ_BLD);
  float*  xdb    = alloc_f32((size_t)MROWS * XDBW);
  float*  delta  = alloc_f32(SZ_BLD);
  __bf16* xbf    = alloc_bf((size_t)MROWS * DIMV);     // current layer input (bf16)
  __bf16* ubf    = alloc_bf(SZ_BLD);
  __bf16* xdbbf  = alloc_bf((size_t)MROWS * XDBW);
  __bf16* ygbf   = alloc_bf(SZ_BLD);
  __bf16* WinBf  = alloc_bf((size_t)NLAY * 2 * DINV * DIMV);
  __bf16* WxBf   = alloc_bf((size_t)NLAY * XDBW * DINV);
  __bf16* WdtBf  = alloc_bf((size_t)NLAY * DINV * DTRV);
  __bf16* WoutBf = alloc_bf((size_t)NLAY * DIMV * DINV);

  auto cvt = [&](const float* src, __bf16* dst, size_t n) {
    cvt_bf16_kernel<<<(unsigned)((n / 4 + 255) / 256), 256, 0, stream>>>(src, dst, (int)n);
  };
  auto launch_gemm = [&](const __bf16* Ap, int lda, const __bf16* Wp, int ldw,
                         float* Cp, __bf16* Cbfp, int ldc, int M, int N, int K,
                         const float* bias, int act) {
    const int tiles = (M / 32) * (N / 32);
    dim3 grid((tiles + 7) / 8), block(256);
    gemm_wmma_bf16<<<grid, block, 0, stream>>>(Ap, lda, Wp, ldw, Cp, Cbfp, ldc,
                                               M, N, K, bias, act);
  };

  // One-time (per call) weight + input conversion to bf16
  cvt(W_in,  WinBf,  (size_t)NLAY * 2 * DINV * DIMV);
  cvt(W_x,   WxBf,   (size_t)NLAY * XDBW * DINV);
  cvt(W_dt,  WdtBf,  (size_t)NLAY * DINV * DTRV);
  cvt(W_out, WoutBf, (size_t)NLAY * DIMV * DINV);
  cvt(x,     xbf,    (size_t)MROWS * DIMV);

  for (int i = 0; i < NLAY; ++i) {
    const __bf16* Wi  = WinBf  + (size_t)i * 2 * DINV * DIMV;
    const __bf16* Wxp = WxBf   + (size_t)i * XDBW * DINV;
    const __bf16* Wdp = WdtBf  + (size_t)i * DINV * DTRV;
    const __bf16* Wop = WoutBf + (size_t)i * DIMV * DINV;
    const float*  cw  = conv_w + (size_t)i * DINV * DCV;
    const float*  cb  = conv_b + (size_t)i * DINV;
    const float*  bdt = b_dt   + (size_t)i * DINV;
    const float*  Al  = A_log  + (size_t)i * DINV * DSTV;
    const float*  Dp  = Dv     + (size_t)i * DINV;
    const bool last = (i == NLAY - 1);

    // xi = x @ W_in[:DIN].T ; z = x @ W_in[DIN:].T      (f32 outputs)
    launch_gemm(xbf, DIMV, Wi,                         DIMV, xi,   nullptr, DINV,
                MROWS, DINV, DIMV, nullptr, 0);
    launch_gemm(xbf, DIMV, Wi + (size_t)DINV * DIMV,   DIMV, zbuf, nullptr, DINV,
                MROWS, DINV, DIMV, nullptr, 0);

    // u = silu(depthwise_conv(xi) + cb)   (f32 + bf16)
    conv_silu_kernel<<<(MROWS * DINV) / 256, 256, 0, stream>>>(xi, cw, cb, ubuf, ubf);

    // xdb = u @ W_x.T   rows: [dt(32) | B(16) | C(16)]   (f32 for scan, bf16 for dt GEMM)
    launch_gemm(ubf, DINV, Wxp, DINV, xdb, xdbbf, XDBW,
                MROWS, XDBW, DINV, nullptr, 0);

    // delta = softplus(dt @ W_dt.T + b_dt)   (dt = first 32 cols of xdbbf, lda=64)
    launch_gemm(xdbbf, XDBW, Wdp, DTRV, delta, nullptr, DINV,
                MROWS, DINV, DTRV, bdt, 1);

    // sequential SSM scan + u*D skip + silu(z) gate -> ygbf (bf16)
    scan_gate_kernel<<<(BSZV * DINV) / 256, 256, 0, stream>>>(
        delta, ubuf, xdb, Al, Dp, zbuf, ygbf);

    // out = yg @ W_out.T : last layer -> f32 d_out, else bf16 x for next layer
    launch_gemm(ygbf, DINV, Wop, DINV,
                last ? (float*)d_out : nullptr,
                last ? nullptr : xbf,
                DIMV, MROWS, DIMV, DINV, nullptr, 0);
  }
}
